// AngleRegressionModel_69028714381436
// MI455X (gfx1250) — matrix operand order repaired and checked by reading, not compiled, not run on hardware
//
#include <hip/hip_runtime.h>
#include <stdint.h>

// ---------------- problem constants ----------------
#define NB     8
#define LLEN   256
#define LREC   224
#define LPEP   32
#define NODE   128
#define PAIRD  64
#define NH     12
#define QKD    32
#define VDIM   32
#define NQP    8
#define NVP    8
#define NLAYER 6
#define OUT_IN 1824   // H*V(384) + H*PAIR(768) + H*NVP*7(672)
#define BLrows (NB*LLEN)          // 2048
#define RINV3  0.57735026918962576f  // sqrt(1/3)

typedef __attribute__((ext_vector_type(16))) __bf16 bf16x16;
typedef __attribute__((ext_vector_type(8)))  float  f32x8;

union AFrag { bf16x16 v; uint4 q[2]; unsigned short u[16]; };

__device__ __forceinline__ unsigned short f2bf(float f) {
  unsigned int u = __float_as_uint(f);
  unsigned int r = (u + 0x7FFFu + ((u >> 16) & 1u)) >> 16;
  return (unsigned short)r;
}

__device__ __forceinline__ float warpRedSum(float v) {
#pragma unroll
  for (int o = 16; o > 0; o >>= 1) v += __shfl_xor(v, o, 32);
  return v;
}
__device__ __forceinline__ float warpRedMax(float v) {
#pragma unroll
  for (int o = 16; o > 0; o >>= 1) v = fmaxf(v, __shfl_xor(v, o, 32));
  return v;
}

// ---------------- generic batched WMMA GEMM ----------------
// C[m,n] = sum_k A[m,k] * B[k,n]  (bf16 in, f32 accum)
// A element (m,k) at A + aOff + m*lda + k          (k contiguous, 16B aligned rows)
// B element (k,n) at B + bOff + k*ldb_k + n*ldb_n
//   bmode==1 : ldb_n==1 (n contiguous)  -> double-buffered LDS transpose, b64 loads
//   bmode==2 : ldb_k==1 (k contiguous)  -> direct global fragment loads, no LDS
// batch offset: z = blockIdx.z; off = (z/inner)*Outer + (z%inner)*Inner
// Out-of-range rows/cols are CLAMPED on load (their products only feed C rows/cols
// that the guarded epilogue never writes), so the hot loop is branch-free.
struct GemmP {
  const unsigned short* A;
  const unsigned short* B;
  float* C;                 // optional f32 out
  unsigned short* Cb;       // optional bf16 out (same layout)
  const float* bias;        // optional per-column bias
  int M, N, K;
  int lda, ldb_k, ldb_n, ldc;
  long long aOuter, aInner, bOuter, bInner, cOuter, cInner;
  int innerCnt;
  int relu;
  int bmode;
};

__global__ __launch_bounds__(256) void gemm_wmma(GemmP p) {
  const int tid  = threadIdx.x;
  const int lane = tid & 31;
  const int wave = tid >> 5;
  const int wm = wave >> 1, wn = wave & 1;            // 4x2 wave tiles -> 64x32 block
  const int z  = blockIdx.z;
  const int zo = z / p.innerCnt, zi = z % p.innerCnt;
  const unsigned short* Ab = p.A + (long long)zo * p.aOuter + (long long)zi * p.aInner;
  const unsigned short* Bb = p.B + (long long)zo * p.bOuter + (long long)zi * p.bInner;
  const long long cOff = (long long)zo * p.cOuter + (long long)zi * p.cInner;
  const int row0  = blockIdx.x * 64 + wm * 16;
  const int colB0 = blockIdx.y * 32;
  const int col0  = colB0 + wn * 16;

  f32x8 acc = {};
  const int mRaw = row0 + (lane & 15);
  const int m    = mRaw < p.M ? mRaw : (p.M - 1);       // clamp (safe, see header)
  const int kb   = (lane < 16) ? 0 : 8;                 // A layout half-select
  const int kb2  = (lane < 16) ? 0 : 16;                // unused: B also uses kb
  const unsigned short* Ap = Ab + m * p.lda + kb;

  if (p.bmode == 2) {
    // ---- direct-global B path (B rows k-contiguous): no LDS, no barriers ----
    const int nRaw = col0 + (lane & 15);
    const int n    = nRaw < p.N ? nRaw : (p.N - 1);
    const unsigned short* Bp = Bb + n * p.ldb_n + kb;
    for (int k0 = 0; k0 < p.K; k0 += 32) {
      AFrag a, bf;
      a.q[0]  = ((const uint4*)(Ap + k0))[0];   // K = kb..kb+7
      a.q[1]  = ((const uint4*)(Ap + k0))[2];   // K = kb+16..kb+23
      bf.q[0] = ((const uint4*)(Bp + k0))[0];   // K = kb..kb+7
      bf.q[1] = ((const uint4*)(Bp + k0))[2];
      acc = __builtin_amdgcn_wmma_f32_16x16x32_bf16(false, a.v, false, bf.v,
                                                    (short)0, acc, false, false);
    }
  } else {
    // ---- double-buffered LDS path (B rows n-contiguous) ----
    __shared__ __align__(16) unsigned short Bs[2][32][32];  // [buf][n][k]
    const int skk = tid >> 3;          // 0..31  (k within tile)
    const int snn = (tid & 7) * 4;     // 0,4,..28 (n group within tile)
    const unsigned short* Bst = Bb + colB0 + snn;           // + (k+skk)*ldb_k
    const int nIn = wn * 16 + (lane & 15);

    // prologue: stage tile 0 into buffer 0
    {
      uint2 raw = *(const uint2*)(Bst + skk * p.ldb_k);
      Bs[0][snn + 0][skk] = (unsigned short)(raw.x);
      Bs[0][snn + 1][skk] = (unsigned short)(raw.x >> 16);
      Bs[0][snn + 2][skk] = (unsigned short)(raw.y);
      Bs[0][snn + 3][skk] = (unsigned short)(raw.y >> 16);
    }
    __syncthreads();

    int buf = 0;
    for (int k0 = 0; k0 < p.K; k0 += 32) {
      // issue next tile's staging load early (branch-free: last step re-reads k=0
      // into the dead buffer; in-slice, values never consumed)
      const int k0n = (k0 + 32 < p.K) ? (k0 + 32) : 0;
      uint2 raw = *(const uint2*)(Bst + (k0n + skk) * p.ldb_k);

      AFrag a, bf;
      a.q[0] = ((const uint4*)(Ap + k0))[0];
      a.q[1] = ((const uint4*)(Ap + k0))[2];
      const unsigned short* bp = &Bs[buf][nIn][kb];
      bf.q[0] = ((const uint4*)bp)[0];
      bf.q[1] = ((const uint4*)bp)[2];
      acc = __builtin_amdgcn_wmma_f32_16x16x32_bf16(false, a.v, false, bf.v,
                                                    (short)0, acc, false, false);

      // transpose-store next tile into the other buffer (after wmma issue)
      const int nb = buf ^ 1;
      Bs[nb][snn + 0][skk] = (unsigned short)(raw.x);
      Bs[nb][snn + 1][skk] = (unsigned short)(raw.x >> 16);
      Bs[nb][snn + 2][skk] = (unsigned short)(raw.y);
      Bs[nb][snn + 3][skk] = (unsigned short)(raw.y >> 16);
      __syncthreads();          // single barrier per step
      buf = nb;
    }
  }

  const int cn = col0 + (lane & 15);
  const int rbase = row0 + ((lane < 16) ? 0 : 8);
  if (cn < p.N) {
    float bias = p.bias ? p.bias[cn] : 0.0f;
#pragma unroll
    for (int r = 0; r < 8; ++r) {
      int cm = rbase + r;
      if (cm < p.M) {
        float v = acc[r] + bias;
        if (p.relu) v = fmaxf(v, 0.0f);
        long long off = cOff + (long long)cm * p.ldc + cn;
        if (p.C)  p.C[off]  = v;
        if (p.Cb) p.Cb[off] = f2bf(v);
      }
    }
  }
}

// ---------------- f32 -> bf16 convert ----------------
__global__ void k_f2bf(const float* __restrict__ src, unsigned short* __restrict__ dst, long long n) {
  long long i = (long long)blockIdx.x * blockDim.x + threadIdx.x;
  if (i < n) dst[i] = f2bf(src[i]);
}

// ---------------- pack Wpb (NLAYER,64,12) -> bf16 (NLAYER,64,16) padded ----------------
__global__ void k_packWpb(const float* __restrict__ src, unsigned short* __restrict__ dst) {
  int idx = blockIdx.x * blockDim.x + threadIdx.x;
  if (idx >= NLAYER * PAIRD * 16) return;
  int c = idx & 15;
  int r = idx >> 4;
  dst[idx] = (c < NH) ? f2bf(src[r * NH + c]) : (unsigned short)0;
}

// ---------------- rigid transform for q/k side ----------------
// Builds fused operand (b,h,l,64): [node*sclN , pts_g*sclS , 0pad8], plus |pts_g|^2.
__global__ void k_rigid_qk(const float* __restrict__ node,   // (BL, H*32)
                           const float* __restrict__ pts,    // (BL, H*24)
                           const float* __restrict__ R,      // (BL, 3,3)
                           const float* __restrict__ t,      // (BL, 3)
                           const float* __restrict__ coef,   // (H) this layer
                           unsigned short* __restrict__ out64, // ((b*H+h)*L + l)*64
                           float* __restrict__ nrm,          // (BL, H)
                           int isQ) {
  int idx = blockIdx.x * blockDim.x + threadIdx.x;
  if (idx >= BLrows * NH) return;
  int h = idx % NH;
  int bl = idx / NH;
  int l = bl % LLEN, b = bl / LLEN;

  float Rm[9];
#pragma unroll
  for (int e = 0; e < 9; ++e) Rm[e] = R[(long long)bl * 9 + e];
  float t0 = t[bl * 3 + 0], t1 = t[bl * 3 + 1], t2 = t[bl * 3 + 2];

  float gamma = log1pf(__expf(coef[h]));
  float sclN = isQ ? (RINV3 / sqrtf((float)QKD)) : 1.0f;
  // logits_spatial = s_h*d2, s_h = -gamma/12 ; cross coeff folded = (-2*s_h*r)
  float sclS = isQ ? (2.0f * gamma * (1.0f / 12.0f) * RINV3) : 1.0f;

  unsigned short* o = out64 + ((long long)(b * NH + h) * LLEN + l) * 64;
  const float* nrow = node + (long long)bl * (NH * 32) + h * 32;
#pragma unroll
  for (int d = 0; d < 32; ++d) o[d] = f2bf(nrow[d] * sclN);

  const float* prow = pts + (long long)bl * (NH * 24) + h * 24;
  float nacc = 0.0f;
#pragma unroll
  for (int pp = 0; pp < NQP; ++pp) {
    float x = prow[pp * 3 + 0], y = prow[pp * 3 + 1], zc = prow[pp * 3 + 2];
    float gx = Rm[0] * x + Rm[1] * y + Rm[2] * zc + t0;
    float gy = Rm[3] * x + Rm[4] * y + Rm[5] * zc + t1;
    float gz = Rm[6] * x + Rm[7] * y + Rm[8] * zc + t2;
    nacc += gx * gx + gy * gy + gz * gz;
    o[32 + pp * 3 + 0] = f2bf(gx * sclS);
    o[32 + pp * 3 + 1] = f2bf(gy * sclS);
    o[32 + pp * 3 + 2] = f2bf(gz * sclS);
  }
#pragma unroll
  for (int d = 56; d < 64; ++d) o[d] = 0;
  nrm[(long long)bl * NH + h] = nacc;
}

// ---------------- rigid transform for v points ----------------
__global__ void k_rigid_v(const float* __restrict__ pts,   // (BL, H*24)
                          const float* __restrict__ R, const float* __restrict__ t,
                          unsigned short* __restrict__ outb) { // (BL, H*24) bf16
  int idx = blockIdx.x * blockDim.x + threadIdx.x;
  if (idx >= BLrows * NH) return;
  int h = idx % NH;
  int bl = idx / NH;
  float Rm[9];
#pragma unroll
  for (int e = 0; e < 9; ++e) Rm[e] = R[(long long)bl * 9 + e];
  float t0 = t[bl * 3 + 0], t1 = t[bl * 3 + 1], t2 = t[bl * 3 + 2];
  const float* prow = pts + (long long)bl * (NH * 24) + h * 24;
  unsigned short* o = outb + (long long)bl * (NH * 24) + h * 24;
#pragma unroll
  for (int pp = 0; pp < NVP; ++pp) {
    float x = prow[pp * 3 + 0], y = prow[pp * 3 + 1], zc = prow[pp * 3 + 2];
    o[pp * 3 + 0] = f2bf(Rm[0] * x + Rm[1] * y + Rm[2] * zc + t0);
    o[pp * 3 + 1] = f2bf(Rm[3] * x + Rm[4] * y + Rm[5] * zc + t1);
    o[pp * 3 + 2] = f2bf(Rm[6] * x + Rm[7] * y + Rm[8] * zc + t2);
  }
}

// ---------------- fused logits assembly + softmax ----------------
// block = (b,i,h), 256 threads over j
__global__ __launch_bounds__(256) void k_softmax(
    const float* __restrict__ lnc,   // ((b*H+h)*L+i)*L + j : node+cross combined (already *r-scaled)
    const float* __restrict__ pb,    // ((b*L+i)*L+j)*16 + h (padded to 16)
    const float* __restrict__ nq,    // (b*L+i)*12 + h
    const float* __restrict__ nk,    // (b*L+j)*12 + h
    const float* __restrict__ coef,  // (H) this layer
    const unsigned char* __restrict__ mask, // (B,L)
    unsigned short* __restrict__ alphab) {  // ((b*L+i)*12+h)*L + j
  __shared__ float sm[8];
  int z = blockIdx.x;
  int h = z % NH;
  int bi = z / NH;
  int i = bi % LLEN, b = bi / LLEN;
  int j = threadIdx.x;
  int lane = j & 31, w = j >> 5;

  float gamma = log1pf(__expf(coef[h]));
  float s_h = -gamma * (1.0f / 12.0f);

  float lv = lnc[((long long)(b * NH + h) * LLEN + i) * LLEN + j];
  float pv = pb[((long long)(b * LLEN + i) * LLEN + j) * 16 + h];
  float nqi = nq[(long long)(b * LLEN + i) * NH + h];
  float nkj = nk[(long long)(b * LLEN + j) * NH + h];
  float logit = lv + RINV3 * pv + RINV3 * s_h * (nqi + nkj);

  bool mi = mask[b * LLEN + i] != 0;
  bool mj = mask[b * LLEN + j] != 0;
  if (!(mi && mj)) logit -= 100000.0f;

  float wmax = warpRedMax(logit);
  if (lane == 0) sm[w] = wmax;
  __syncthreads();
  float mx = sm[0];
#pragma unroll
  for (int e = 1; e < 8; ++e) mx = fmaxf(mx, sm[e]);
  __syncthreads();

  float ex = __expf(logit - mx);
  float wsum = warpRedSum(ex);
  if (lane == 0) sm[w] = wsum;
  __syncthreads();
  float tot = sm[0];
#pragma unroll
  for (int e = 1; e < 8; ++e) tot += sm[e];

  float a = mi ? (ex / tot) : 0.0f;
  alphab[((long long)bi * NH + h) * LLEN + j] = f2bf(a);
}

// ---------------- back-project aggregated points, write spatial feats ----------------
__global__ void k_fp(const float* __restrict__ aggr,  // ((b*H+h)*L+i)*24
                     const float* __restrict__ R, const float* __restrict__ t,
                     float* __restrict__ featAll) {   // (BL,1824)
  int idx = blockIdx.x * blockDim.x + threadIdx.x;
  if (idx >= BLrows * NH * NVP) return;
  int pp = idx & 7;
  int rest = idx >> 3;
  int h = rest % NH;
  int bl = rest / NH;
  int i = bl % LLEN, b = bl / LLEN;
  const float* a = aggr + ((long long)(b * NH + h) * LLEN + i) * 24 + pp * 3;
  float dx = a[0] - t[bl * 3 + 0];
  float dy = a[1] - t[bl * 3 + 1];
  float dz = a[2] - t[bl * 3 + 2];
  const float* Rm = R + (long long)bl * 9;
  float fx = Rm[0] * dx + Rm[3] * dy + Rm[6] * dz;   // R^T * d
  float fy = Rm[1] * dx + Rm[4] * dy + Rm[7] * dz;
  float fz = Rm[2] * dx + Rm[5] * dy + Rm[8] * dz;
  float fd = sqrtf(fx * fx + fy * fy + fz * fz);
  float inv = 1.0f / (fd + 1e-4f);
  float* row = featAll + (long long)bl * OUT_IN;
  row[1152 + h * 24 + pp * 3 + 0] = fx;
  row[1152 + h * 24 + pp * 3 + 1] = fy;
  row[1152 + h * 24 + pp * 3 + 2] = fz;
  row[1440 + h * 8 + pp] = fd;
  row[1536 + h * 24 + pp * 3 + 0] = fx * inv;
  row[1536 + h * 24 + pp * 3 + 1] = fy * inv;
  row[1536 + h * 24 + pp * 3 + 2] = fz * inv;
}

// ---------------- residual + LayerNorm (writes f32 + bf16) ----------------
__global__ __launch_bounds__(128) void k_addln(
    const float* __restrict__ xin, const float* __restrict__ delta,
    const unsigned char* __restrict__ mask,  // null for ln2
    const float* __restrict__ g, const float* __restrict__ bta,
    float* __restrict__ xout, unsigned short* __restrict__ xbout) {
  __shared__ float sm[4];
  int row = blockIdx.x, c = threadIdx.x;
  int lane = c & 31, w = c >> 5;
  float d = delta[(long long)row * NODE + c];
  if (mask) d *= mask[row] ? 1.0f : 0.0f;
  float v = xin[(long long)row * NODE + c] + d;
  float s = warpRedSum(v);
  if (lane == 0) sm[w] = s;
  __syncthreads();
  float mean = (sm[0] + sm[1] + sm[2] + sm[3]) * (1.0f / NODE);
  float dv = v - mean;
  __syncthreads();
  float s2 = warpRedSum(dv * dv);
  if (lane == 0) sm[w] = s2;
  __syncthreads();
  float var = (sm[0] + sm[1] + sm[2] + sm[3]) * (1.0f / NODE);
  float o = dv * rsqrtf(var + 1e-5f) * g[c] + bta[c];
  xout[(long long)row * NODE + c] = o;
  xbout[(long long)row * NODE + c] = f2bf(o);
}

// ---------------- final angle regression ----------------
__global__ void k_reg(const float* __restrict__ x, const float* __restrict__ Wreg,
                      const float* __restrict__ breg, float* __restrict__ out) {
  int idx = blockIdx.x * blockDim.x + threadIdx.x;
  if (idx >= NB * LPEP * 4) return;
  int o = idx & 3;
  int rest = idx >> 2;
  int lp = rest & 31;
  int b = rest >> 5;
  const float* row = x + (long long)(b * LLEN + LREC + lp) * NODE;
  float acc = breg[o];
#pragma unroll 4
  for (int c = 0; c < NODE; ++c) acc += row[c] * Wreg[c * 4 + o];
  out[idx] = acc;
}

// ---------------- host-side helpers ----------------
static inline void gemm(hipStream_t s, const unsigned short* A, const unsigned short* B,
                        float* C, unsigned short* Cb, const float* bias,
                        int M, int N, int K, int lda, int ldbk, int ldbn, int ldc,
                        long long aO, long long aI, long long bO, long long bI,
                        long long cO, long long cI, int inner, int batches, int relu) {
  int bmode = (ldbk == 1) ? 2 : 1;   // ldbn==1 for all bmode-1 calls
  GemmP p{A, B, C, Cb, bias, M, N, K, lda, ldbk, ldbn, ldc, aO, aI, bO, bI, cO, cI, inner, relu, bmode};
  dim3 grid((M + 63) / 64, (N + 31) / 32, batches);
  hipLaunchKernelGGL(gemm_wmma, grid, dim3(256), 0, s, p);
}

static inline void conv(hipStream_t s, const float* src, unsigned short* dst, long long n) {
  hipLaunchKernelGGL(k_f2bf, dim3((unsigned)((n + 255) / 256)), dim3(256), 0, s, src, dst, n);
}

extern "C" void kernel_launch(void* const* d_in, const int* in_sizes, int n_in,
                              void* d_out, int out_size, void* d_ws, size_t ws_size,
                              hipStream_t stream) {
  const float* rot  = (const float*)d_in[0];
  const float* pos  = (const float*)d_in[1];
  const float* resf = (const float*)d_in[2];
  const float* zf   = (const float*)d_in[3];
  const unsigned char* mask = (const unsigned char*)d_in[4];
  const float* Wq   = (const float*)d_in[5];
  const float* Wk   = (const float*)d_in[6];
  const float* Wv   = (const float*)d_in[7];
  const float* Wpb  = (const float*)d_in[8];
  const float* coef = (const float*)d_in[9];
  const float* Wqp  = (const float*)d_in[10];
  const float* Wkp  = (const float*)d_in[11];
  const float* Wvp  = (const float*)d_in[12];
  const float* Wo   = (const float*)d_in[13];
  const float* bo   = (const float*)d_in[14];
  const float* ln1g = (const float*)d_in[15];
  const float* ln1b = (const float*)d_in[16];
  const float* w1   = (const float*)d_in[17];
  const float* b1   = (const float*)d_in[18];
  const float* w2   = (const float*)d_in[19];
  const float* b2   = (const float*)d_in[20];
  const float* w3   = (const float*)d_in[21];
  const float* b3   = (const float*)d_in[22];
  const float* ln2g = (const float*)d_in[23];
  const float* ln2b = (const float*)d_in[24];
  const float* Wreg = (const float*)d_in[25];
  const float* breg = (const float*)d_in[26];

  // bump allocator on workspace (256B aligned slices; clamped-tile overreads of a
  // few tens of bytes always land in the next slice's padding)
  char* base = (char*)d_ws;
  size_t off = 0;
  auto alloc = [&](size_t bytes) -> void* {
    void* p = base + off;
    off = (off + bytes + 255) & ~(size_t)255;
    return p;
  };

  const long long ZN = (long long)NB * LLEN * LLEN * PAIRD;          // 33.5M
  unsigned short* zb    = (unsigned short*)alloc(ZN * 2);
  unsigned short* wq_b  = (unsigned short*)alloc((size_t)NLAYER * NODE * 384 * 2);
  unsigned short* wk_b  = (unsigned short*)alloc((size_t)NLAYER * NODE * 384 * 2);
  unsigned short* wv_b  = (unsigned short*)alloc((size_t)NLAYER * NODE * 384 * 2);
  unsigned short* wqp_b = (unsigned short*)alloc((size_t)NLAYER * NODE * 288 * 2);
  unsigned short* wkp_b = (unsigned short*)alloc((size_t)NLAYER * NODE * 288 * 2);
  unsigned short* wvp_b = (unsigned short*)alloc((size_t)NLAYER * NODE * 288 * 2);
  unsigned short* wpb_b = (unsigned short*)alloc((size_t)NLAYER * PAIRD * 16 * 2 + 256); // padded N=16 + slack
  unsigned short* wo_b  = (unsigned short*)alloc((size_t)NLAYER * OUT_IN * NODE * 2);
  unsigned short* w1_b  = (unsigned short*)alloc((size_t)NLAYER * NODE * NODE * 2);
  unsigned short* w2_b  = (unsigned short*)alloc((size_t)NLAYER * NODE * NODE * 2);
  unsigned short* w3_b  = (unsigned short*)alloc((size_t)NLAYER * NODE * NODE * 2);

  float*          x    = (float*)alloc((size_t)BLrows * NODE * 4);
  unsigned short* xb   = (unsigned short*)alloc((size_t)BLrows * NODE * 2);
  float*          qf   = (float*)alloc((size_t)BLrows * 384 * 4);
  float*          kf   = (float*)alloc((size_t)BLrows * 384 * 4);
  unsigned short* vb   = (unsigned short*)alloc((size_t)BLrows * 384 * 2);
  float*          qpf  = (float*)alloc((size_t)BLrows * 288 * 4);
  float*          kpf  = (float*)alloc((size_t)BLrows * 288 * 4);
  float*          vpf  = (float*)alloc((size_t)BLrows * 288 * 4);
  unsigned short* A64  = (unsigned short*)alloc((size_t)NB * NH * LLEN * 64 * 2);
  unsigned short* B64  = (unsigned short*)alloc((size_t)NB * NH * LLEN * 64 * 2);
  float*          nq   = (float*)alloc((size_t)BLrows * NH * 4);
  float*          nk   = (float*)alloc((size_t)BLrows * NH * 4);
  unsigned short* vpgb = (unsigned short*)alloc((size_t)BLrows * 288 * 2);
  float*          pbf  = (float*)alloc((size_t)NB * LLEN * LLEN * 16 * 4);   // 33 MB (N padded 16)
  float*          lncf = (float*)alloc((size_t)NB * NH * LLEN * LLEN * 4);   // 25 MB
  unsigned short* alphab = (unsigned short*)alloc((size_t)NB * LLEN * NH * LLEN * 2 + 4096);
  float*          aggrf  = (float*)alloc((size_t)NB * NH * LLEN * 24 * 4);
  float*          featAll  = (float*)alloc((size_t)BLrows * OUT_IN * 4);
  unsigned short* featAllb = (unsigned short*)alloc((size_t)BLrows * OUT_IN * 2);
  float*          yf   = (float*)alloc((size_t)BLrows * NODE * 4);
  unsigned short* h1b  = (unsigned short*)alloc((size_t)BLrows * NODE * 2);
  unsigned short* h2b  = (unsigned short*)alloc((size_t)BLrows * NODE * 2);
  float*          h3f  = (float*)alloc((size_t)BLrows * NODE * 4 + 256);
  (void)ws_size; (void)in_sizes; (void)n_in; (void)out_size;

  // one-time conversions (bf16 copies of pair tensor + all weights)
  conv(stream, zf, zb, ZN);
  conv(stream, Wq,  wq_b,  (long long)NLAYER * NODE * 384);
  conv(stream, Wk,  wk_b,  (long long)NLAYER * NODE * 384);
  conv(stream, Wv,  wv_b,  (long long)NLAYER * NODE * 384);
  conv(stream, Wqp, wqp_b, (long long)NLAYER * NODE * 288);
  conv(stream, Wkp, wkp_b, (long long)NLAYER * NODE * 288);
  conv(stream, Wvp, wvp_b, (long long)NLAYER * NODE * 288);
  hipLaunchKernelGGL(k_packWpb, dim3((NLAYER * PAIRD * 16 + 255) / 256), dim3(256), 0, stream,
                     Wpb, wpb_b);
  conv(stream, Wo,  wo_b,  (long long)NLAYER * OUT_IN * NODE);
  conv(stream, w1,  w1_b,  (long long)NLAYER * NODE * NODE);
  conv(stream, w2,  w2_b,  (long long)NLAYER * NODE * NODE);
  conv(stream, w3,  w3_b,  (long long)NLAYER * NODE * NODE);

  hipMemcpyAsync(x, resf, (size_t)BLrows * NODE * 4, hipMemcpyDeviceToDevice, stream);
  conv(stream, x, xb, (long long)BLrows * NODE);

  const int BH = NB * NH;  // 96
  for (int l = 0; l < NLAYER; ++l) {
    const unsigned short* wql  = wq_b  + (size_t)l * NODE * 384;
    const unsigned short* wkl  = wk_b  + (size_t)l * NODE * 384;
    const unsigned short* wvl  = wv_b  + (size_t)l * NODE * 384;
    const unsigned short* wqpl = wqp_b + (size_t)l * NODE * 288;
    const unsigned short* wkpl = wkp_b + (size_t)l * NODE * 288;
    const unsigned short* wvpl = wvp_b + (size_t)l * NODE * 288;
    const unsigned short* wpbl = wpb_b + (size_t)l * PAIRD * 16;
    const unsigned short* wol  = wo_b  + (size_t)l * OUT_IN * NODE;
    const unsigned short* w1l  = w1_b  + (size_t)l * NODE * NODE;
    const unsigned short* w2l  = w2_b  + (size_t)l * NODE * NODE;
    const unsigned short* w3l  = w3_b  + (size_t)l * NODE * NODE;
    const float* coefl = coef + (size_t)l * NH;

    // projections: x @ W
    gemm(stream, xb, wql,  qf,   nullptr, nullptr, BLrows, 384, NODE, NODE, 384, 1, 384, 0,0,0,0,0,0, 1, 1, 0);
    gemm(stream, xb, wkl,  kf,   nullptr, nullptr, BLrows, 384, NODE, NODE, 384, 1, 384, 0,0,0,0,0,0, 1, 1, 0);
    gemm(stream, xb, wvl,  nullptr, vb,   nullptr, BLrows, 384, NODE, NODE, 384, 1, 384, 0,0,0,0,0,0, 1, 1, 0);
    gemm(stream, xb, wqpl, qpf,  nullptr, nullptr, BLrows, 288, NODE, NODE, 288, 1, 288, 0,0,0,0,0,0, 1, 1, 0);
    gemm(stream, xb, wkpl, kpf,  nullptr, nullptr, BLrows, 288, NODE, NODE, 288, 1, 288, 0,0,0,0,0,0, 1, 1, 0);
    gemm(stream, xb, wvpl, vpf,  nullptr, nullptr, BLrows, 288, NODE, NODE, 288, 1, 288, 0,0,0,0,0,0, 1, 1, 0);

    // rigid transforms -> fused 64-wide operands + norms; v points bf16
    hipLaunchKernelGGL(k_rigid_qk, dim3((BLrows * NH + 255) / 256), dim3(256), 0, stream,
                       qf, qpf, rot, pos, coefl, A64, nq, 1);
    hipLaunchKernelGGL(k_rigid_qk, dim3((BLrows * NH + 255) / 256), dim3(256), 0, stream,
                       kf, kpf, rot, pos, coefl, B64, nk, 0);
    hipLaunchKernelGGL(k_rigid_v, dim3((BLrows * NH + 255) / 256), dim3(256), 0, stream,
                       vpf, rot, pos, vpgb);

    // pair bias: z @ Wpb  (M = B*L*L, K=64, N=12 padded to 16)
    gemm(stream, zb, wpbl, pbf, nullptr, nullptr, NB * LLEN * LLEN, NH, PAIRD,
         PAIRD, 16, 1, 16, 0,0,0,0,0,0, 1, 1, 0);

    // fused node+spatial-cross logits: per (b,h): A64(L,64) x B64^T -> (L,L)  [bmode 2]
    gemm(stream, A64, B64, lncf, nullptr, nullptr, LLEN, LLEN, 64,
         64, 1, 64, LLEN,
         (long long)LLEN * 64, 0, (long long)LLEN * 64, 0, (long long)LLEN * LLEN, 0,
         1, BH, 0);

    // softmax -> alpha (bf16)
    hipLaunchKernelGGL(k_softmax, dim3(NB * LLEN * NH), dim3(256), 0, stream,
                       lncf, pbf, nq, nk, coefl, mask, alphab);

    // feat_node: per (b,h): alpha(L,L) @ v(L,32) -> featAll[:, h*32 : h*32+32]
    gemm(stream, alphab, vb, featAll, nullptr, nullptr, LLEN, VDIM, LLEN,
         NH * LLEN, 384, 1, OUT_IN,
         (long long)LLEN * NH * LLEN, LLEN, (long long)LLEN * 384, VDIM,
         (long long)LLEN * OUT_IN, VDIM, NH, BH, 0);

    // feat_pair: per (b,i): alpha_h(12,L) @ z(L,64) -> featAll[:, 384 + h*64 + d]
    gemm(stream, alphab, zb, featAll + 384, nullptr, nullptr, NH, PAIRD, LLEN,
         LLEN, PAIRD, 1, PAIRD,
         (long long)NH * LLEN, 0, (long long)LLEN * PAIRD, 0, (long long)OUT_IN, 0,
         1, BLrows, 0);

    // aggr: per (b,h): alpha(L,L) @ vp_g(L,24)
    gemm(stream, alphab, vpgb, aggrf, nullptr, nullptr, LLEN, 24, LLEN,
         NH * LLEN, 288, 1, 24,
         (long long)LLEN * NH * LLEN, LLEN, (long long)LLEN * 288, 24,
         (long long)NH * LLEN * 24, (long long)LLEN * 24, NH, BH, 0);

    // back-project, spatial features into featAll[:, 1152:1824]
    hipLaunchKernelGGL(k_fp, dim3((BLrows * NH * NVP + 255) / 256), dim3(256), 0, stream,
                       aggrf, rot, pos, featAll);

    // output projection: featAll @ Wo + bo
    conv(stream, featAll, featAllb, (long long)BLrows * OUT_IN);
    gemm(stream, featAllb, wol, yf, nullptr, bo + (size_t)l * NODE,
         BLrows, NODE, OUT_IN, OUT_IN, NODE, 1, NODE, 0,0,0,0,0,0, 1, 1, 0);

    // x = LN(x + y*mask)
    hipLaunchKernelGGL(k_addln, dim3(BLrows), dim3(NODE), 0, stream,
                       x, yf, mask, ln1g + (size_t)l * NODE, ln1b + (size_t)l * NODE, x, xb);

    // MLP
    gemm(stream, xb,  w1l, nullptr, h1b, b1 + (size_t)l * NODE, BLrows, NODE, NODE,
         NODE, NODE, 1, NODE, 0,0,0,0,0,0, 1, 1, 1);
    gemm(stream, h1b, w2l, nullptr, h2b, b2 + (size_t)l * NODE, BLrows, NODE, NODE,
         NODE, NODE, 1, NODE, 0,0,0,0,0,0, 1, 1, 1);
    gemm(stream, h2b, w3l, h3f, nullptr, b3 + (size_t)l * NODE, BLrows, NODE, NODE,
         NODE, NODE, 1, NODE, 0,0,0,0,0,0, 1, 1, 0);

    // x = LN(x + h)
    hipLaunchKernelGGL(k_addln, dim3(BLrows), dim3(NODE), 0, stream,
                       x, h3f, nullptr, ln2g + (size_t)l * NODE, ln2b + (size_t)l * NODE, x, xb);
  }

  // final regression on peptide rows
  hipLaunchKernelGGL(k_reg, dim3((NB * LPEP * 4 + 255) / 256), dim3(256), 0, stream,
                     x, Wreg, breg, (float*)d_out);
}
